// HilbertTransformerBlock_58926951301486
// MI455X (gfx1250) — compile-verified, matrix operand order checked by reading, not run on hardware
//
#include <hip/hip_runtime.h>
#include <hip/hip_bf16.h>

typedef unsigned int   u32;
typedef unsigned short u16;

typedef __attribute__((ext_vector_type(16))) __bf16 v16bf;
typedef __attribute__((ext_vector_type(8)))  float  v8f;
typedef __attribute__((ext_vector_type(4)))  u32    v4u;
typedef __attribute__((ext_vector_type(8)))  u32    v8u;
typedef __attribute__((ext_vector_type(4)))  int    i4;

#define BN     4
#define NTOK   4096
#define DM     1024
#define NHEAD  16
#define DH     64
#define SEG    256
#define NSEG   16          // NTOK / SEG
#define SR     128         // SEG / R, R = 2
#define ASTR   40          // LDS row stride in bf16 elems (32 data + 8 pad, 80B = 16B-aligned)

#if __has_builtin(__builtin_amdgcn_global_load_async_to_lds_b128) && \
    __has_builtin(__builtin_amdgcn_s_wait_asynccnt)
#define USE_ASYNC_LDS 1
#else
#define USE_ASYNC_LDS 0
#endif

// ---------- bf16 helpers ----------
__device__ __forceinline__ u16 f2bf(float f) {
    u32 u = __builtin_bit_cast(u32, f);
    u32 r = (u + 0x7FFFu + ((u >> 16) & 1u)) >> 16;   // round-to-nearest-even
    return (u16)r;
}
__device__ __forceinline__ float bf2f(u16 h) {
    u32 u = ((u32)h) << 16;
    return __builtin_bit_cast(float, u);
}

// 16-byte global -> LDS copy: async (ASYNCcnt-tracked) path on gfx1250, staged fallback otherwise
__device__ __forceinline__ void cp16_g2l(const v4u* g, u16* l) {
#if USE_ASYNC_LDS
    __builtin_amdgcn_global_load_async_to_lds_b128(
        (__attribute__((address_space(1))) i4*)g,
        (__attribute__((address_space(3))) i4*)l, 0, 0);
#else
    *(v4u*)l = *g;
#endif
}

// ---------- weight transpose + fp32 -> bf16 convert:  wT[n*K + k] = bf16(w[k*N + n]) ----------
__global__ void convT_kernel(const float* __restrict__ w, u16* __restrict__ wT, int K, int N) {
    __shared__ float tile[32][33];
    int kb = blockIdx.y * 32, nb = blockIdx.x * 32;
    int tx = threadIdx.x, ty = threadIdx.y;   // block (32, 8)
    #pragma unroll
    for (int r = 0; r < 4; ++r)
        tile[ty + r * 8][tx] = w[(size_t)(kb + ty + r * 8) * N + (nb + tx)];
    __syncthreads();
    #pragma unroll
    for (int r = 0; r < 4; ++r)
        wT[(size_t)(nb + ty + r * 8) * K + (kb + tx)] = f2bf(tile[tx][ty + r * 8]);
}

// ---------- LayerNorm (optionally fused with Hilbert gather), fp32 in -> bf16 out ----------
__global__ void ln_kernel(const float* __restrict__ x, const float* __restrict__ g,
                          const float* __restrict__ bta, const int* __restrict__ perm,
                          u16* __restrict__ out, int usePerm) {
    int row = blockIdx.x;                     // 0 .. B*N-1 (destination row, permuted domain)
    int b = row >> 12, i = row & (NTOK - 1);
    int src = usePerm ? ((b << 12) + perm[i]) : row;
    const float* xr = x + (size_t)src * DM;
    int tid = threadIdx.x;                    // 256 threads, 4 elems each
    float v[4], s = 0.f, ss = 0.f;
    #pragma unroll
    for (int t = 0; t < 4; ++t) {
        v[t] = xr[tid + t * 256];
        s += v[t]; ss += v[t] * v[t];
    }
    __shared__ float rs_[256], rss_[256];
    rs_[tid] = s; rss_[tid] = ss;
    __syncthreads();
    for (int o = 128; o > 0; o >>= 1) {
        if (tid < o) { rs_[tid] += rs_[tid + o]; rss_[tid] += rss_[tid + o]; }
        __syncthreads();
    }
    float mu  = rs_[0]  * (1.f / DM);
    float var = rss_[0] * (1.f / DM) - mu * mu;
    float rin = rsqrtf(var + 1e-5f);
    #pragma unroll
    for (int t = 0; t < 4; ++t) {
        int d = tid + t * 256;
        out[(size_t)row * DM + d] = f2bf((v[t] - mu) * rin * g[d] + bta[d]);
    }
}

// ---------- bf16 WMMA GEMM with async-LDS staging ----------
// Block: 256 threads = 8 waves (4x2), block tile 256x128, wave tile 64x64.
// K stepped by 32, double-buffered LDS (60 KB static).
// EPI 0: out bf16 = acc + bias
// EPI 1: out bf16 = gelu(acc + bias)
// EPI 2: out f32  at scatter row (b*NTOK + perm[i]) = resid[trow] + acc + bias   (o-proj un-permute)
// EPI 3: out f32  = resid[row] + acc + bias                                       (final residual)
template <int EPI>
__global__ __launch_bounds__(256) void gemm_bf16_wmma(
    const u16* __restrict__ A, const u16* __restrict__ BT,
    const float* __restrict__ bias, const float* __restrict__ resid,
    const int* __restrict__ perm, void* __restrict__ outp,
    int M, int Ncols, int K) {

    __shared__ u16 As[2][256 * ASTR];   // 40960 B
    __shared__ u16 Bs[2][128 * ASTR];   // 20480 B

    const int tid  = threadIdx.x;
    const int lane = tid & 31;
    const int wave = tid >> 5;               // 8 waves
    const int wrow = wave >> 1;              // 0..3  -> 64-row strip
    const int wcol = wave & 1;               // 0..1  -> 64-col strip
    const int rowBlk = blockIdx.y * 256;
    const int colBlk = blockIdx.x * 128;
    const int lm   = lane & 15;
    const int koff = (lane < 16) ? 0 : 8;    // 16-bit fragment K-halves per ISA layout

    v8f acc[4][4];
    #pragma unroll
    for (int mt = 0; mt < 4; ++mt)
        #pragma unroll
        for (int nt = 0; nt < 4; ++nt) acc[mt][nt] = (v8f)0.f;

    // stage loader: copies the 256x32 A tile and 128x32 B tile into LDS buffer `buf`
    auto stage = [&](int buf, int k0) {
        #pragma unroll
        for (int i = 0; i < 4; ++i) {                       // A: 1024 16B chunks / 256 thr
            int c = tid + i * 256;
            int row = c >> 2, cc = (c & 3) * 8;
            cp16_g2l((const v4u*)(A + (size_t)(rowBlk + row) * K + k0 + cc),
                     &As[buf][row * ASTR + cc]);
        }
        #pragma unroll
        for (int i = 0; i < 2; ++i) {                       // B: 512 16B chunks / 256 thr
            int c = tid + i * 256;
            int row = c >> 2, cc = (c & 3) * 8;
            cp16_g2l((const v4u*)(BT + (size_t)(colBlk + row) * K + k0 + cc),
                     &Bs[buf][row * ASTR + cc]);
        }
    };

    const int nst = K >> 5;
    stage(0, 0);
    for (int s = 0; s < nst; ++s) {
        if (s + 1 < nst) {
            stage((s + 1) & 1, (s + 1) << 5);
#if USE_ASYNC_LDS
            __builtin_amdgcn_s_wait_asynccnt(6);   // 6 outstanding = next stage only
#endif
        } else {
#if USE_ASYNC_LDS
            __builtin_amdgcn_s_wait_asynccnt(0);
#endif
        }
        __syncthreads();                            // stage s visible to all waves

        const int buf = s & 1;
        v16bf afr[4], bfr[4];
        #pragma unroll
        for (int mt = 0; mt < 4; ++mt) {
            const u16* ar = &As[buf][(wrow * 64 + mt * 16 + lm) * ASTR + koff];
            v8u f; f.lo = *(const v4u*)ar; f.hi = *(const v4u*)(ar + 16);
            afr[mt] = __builtin_bit_cast(v16bf, f);
        }
        #pragma unroll
        for (int nt = 0; nt < 4; ++nt) {
            const u16* br = &Bs[buf][(wcol * 64 + nt * 16 + lm) * ASTR + koff];
            v8u f; f.lo = *(const v4u*)br; f.hi = *(const v4u*)(br + 16);
            bfr[nt] = __builtin_bit_cast(v16bf, f);
        }
        #pragma unroll
        for (int mt = 0; mt < 4; ++mt)
            #pragma unroll
            for (int nt = 0; nt < 4; ++nt)
                acc[mt][nt] = __builtin_amdgcn_wmma_f32_16x16x32_bf16(
                    false, afr[mt], false, bfr[nt], (short)0, acc[mt][nt], false, false);

        __syncthreads();                            // protect buffer reuse (s+2 writes buf)
    }

    // ---- epilogue: VGPR j of v8f -> row j + (lane<16 ? 0 : 8); col = lane%16 ----
    const int rsel = (lane < 16) ? 0 : 8;
    #pragma unroll
    for (int mt = 0; mt < 4; ++mt) {
        #pragma unroll
        for (int nt = 0; nt < 4; ++nt) {
            int col = colBlk + wcol * 64 + nt * 16 + lm;
            float bv = bias[col];
            #pragma unroll
            for (int j = 0; j < 8; ++j) {
                int row = rowBlk + wrow * 64 + mt * 16 + j + rsel;
                float v = acc[mt][nt][j] + bv;
                if constexpr (EPI == 0) {
                    ((u16*)outp)[(size_t)row * Ncols + col] = f2bf(v);
                } else if constexpr (EPI == 1) {
                    v = 0.5f * v * (1.f + erff(v * 0.70710678118654752f));
                    ((u16*)outp)[(size_t)row * Ncols + col] = f2bf(v);
                } else if constexpr (EPI == 2) {
                    int b = row >> 12, i = row & (NTOK - 1);
                    size_t trow = (size_t)(b << 12) + perm[i];
                    ((float*)outp)[trow * Ncols + col] = resid[trow * Ncols + col] + v;
                } else {
                    ((float*)outp)[(size_t)row * Ncols + col] =
                        resid[(size_t)row * Ncols + col] + v;
                }
            }
        }
    }
}

// ---------- dilated (Hilbert-segment) attention, one block per (b, seg, head), 128 threads ----------
__global__ __launch_bounds__(128) void attn_kernel(const u16* __restrict__ qkv,
                                                   u16* __restrict__ out) {
    int fid = blockIdx.x;                 // b*256 + g*16 + h
    int h = fid & 15, g = (fid >> 4) & 15, b = fid >> 8;
    int parity = h & 1;                   // offs = h % R, R = 2
    int j = threadIdx.x;                  // q-row 0..127

    __shared__ v4u KsV[SR * 8], VsV[SR * 8];     // 128 x 64 bf16 each (16 KB each)
    const u16* Ks = (const u16*)KsV;
    const u16* Vs = (const u16*)VsV;

    int p = parity + 2 * j;
    size_t tok = ((size_t)(b * NSEG + g)) * SEG + p;
    const u16* qrow = qkv + tok * (3 * DM) + h * DH;
    const u16* krow = qrow + DM;
    const u16* vrow = qrow + 2 * DM;

    #pragma unroll
    for (int t = 0; t < 8; ++t) {
        KsV[j * 8 + t] = ((const v4u*)krow)[t];
        VsV[j * 8 + t] = ((const v4u*)vrow)[t];
    }
    __syncthreads();

    float q[DH];
    #pragma unroll
    for (int d = 0; d < DH; ++d) q[d] = bf2f(qrow[d]);

    // flash-style online softmax over the 128 keys
    float mx = -3.0e38f, sum = 0.f, o[DH];
    #pragma unroll
    for (int d = 0; d < DH; ++d) o[d] = 0.f;

    for (int kk = 0; kk < SR; ++kk) {
        float s = 0.f;
        #pragma unroll
        for (int d = 0; d < DH; ++d) s += q[d] * bf2f(Ks[kk * DH + d]);
        s *= 0.125f;                                   // 1/sqrt(DH)
        float nm   = fmaxf(mx, s);
        float corr = __expf(mx - nm);
        float w    = __expf(s - nm);
        sum = sum * corr + w;
        #pragma unroll
        for (int d = 0; d < DH; ++d) o[d] = o[d] * corr + w * bf2f(Vs[kk * DH + d]);
        mx = nm;
    }
    float inv = 1.f / sum;

    u16* orow = out + tok * DM + h * DH;
    #pragma unroll
    for (int d = 0; d < DH; ++d) orow[d] = f2bf(o[d] * inv);

    // rows of opposite parity receive zeros for this head
    size_t tok2 = ((size_t)(b * NSEG + g)) * SEG + ((parity ^ 1) + 2 * j);
    u16* zrow = out + tok2 * DM + h * DH;
    #pragma unroll
    for (int d = 0; d < DH; d += 8) *(v4u*)(zrow + d) = (v4u)0u;
}

// ---------- host-side orchestration ----------
extern "C" void kernel_launch(void* const* d_in, const int* in_sizes, int n_in,
                              void* d_out, int out_size, void* d_ws, size_t ws_size,
                              hipStream_t stream) {
    const float* x      = (const float*)d_in[0];
    const float* ln1_g  = (const float*)d_in[1];
    const float* ln1_b  = (const float*)d_in[2];
    const float* w_qkv  = (const float*)d_in[3];
    const float* b_qkv  = (const float*)d_in[4];
    const float* w_o    = (const float*)d_in[5];
    const float* b_o    = (const float*)d_in[6];
    const float* ln2_g  = (const float*)d_in[7];
    const float* ln2_b  = (const float*)d_in[8];
    const float* w1     = (const float*)d_in[9];
    const float* b1     = (const float*)d_in[10];
    const float* w2     = (const float*)d_in[11];
    const float* b2     = (const float*)d_in[12];
    const int*   perm   = (const int*)d_in[13];

    const int Mrows = BN * NTOK;             // 16384
    char* w = (char*)d_ws;
    u16* wqkvT = (u16*)w;                    w += (size_t)(3 * DM) * DM * 2;       // 6.0 MB
    u16* woT   = (u16*)w;                    w += (size_t)DM * DM * 2;             // 2.0 MB
    u16* w1T   = (u16*)w;                    w += (size_t)(4 * DM) * DM * 2;       // 8.0 MB
    u16* w2T   = (u16*)w;                    w += (size_t)DM * (4 * DM) * 2;       // 8.0 MB
    u16* xp_ln = (u16*)w;                    w += (size_t)Mrows * DM * 2;          // 32 MB (reused as h2)
    u16* qkv   = (u16*)w;                    w += (size_t)Mrows * (3 * DM) * 2;    // 96 MB
    u16* attno = (u16*)w;                    w += (size_t)Mrows * DM * 2;          // 32 MB
    float* x2  = (float*)w;                  w += (size_t)Mrows * DM * 4;          // 64 MB
    u16* mid   = qkv;                        // aliases qkv+attno: 16384 x 4096 bf16 exactly
    u16* h2    = xp_ln;                      // reuse after QKV GEMM consumed it

    dim3 tb(32, 8);
    // 1) weights -> bf16, transposed
    convT_kernel<<<dim3((3 * DM) / 32, DM / 32), tb, 0, stream>>>(w_qkv, wqkvT, DM, 3 * DM);
    convT_kernel<<<dim3(DM / 32, DM / 32),       tb, 0, stream>>>(w_o,   woT,   DM, DM);
    convT_kernel<<<dim3((4 * DM) / 32, DM / 32), tb, 0, stream>>>(w1,    w1T,   DM, 4 * DM);
    convT_kernel<<<dim3(DM / 32, (4 * DM) / 32), tb, 0, stream>>>(w2,    w2T,   4 * DM, DM);

    // 2) LN1 fused with Hilbert gather
    ln_kernel<<<Mrows, 256, 0, stream>>>(x, ln1_g, ln1_b, perm, xp_ln, 1);

    // 3) QKV GEMM (bf16 WMMA): [16384 x 1024] x [1024 x 3072]
    gemm_bf16_wmma<0><<<dim3((3 * DM) / 128, Mrows / 256), 256, 0, stream>>>(
        xp_ln, wqkvT, b_qkv, nullptr, nullptr, qkv, Mrows, 3 * DM, DM);

    // 4) dilated attention
    attn_kernel<<<BN * NSEG * NHEAD, 128, 0, stream>>>(qkv, attno);

    // 5) O-proj GEMM + inverse-perm scatter + residual -> x2 (fp32)
    gemm_bf16_wmma<2><<<dim3(DM / 128, Mrows / 256), 256, 0, stream>>>(
        attno, woT, b_o, x, perm, x2, Mrows, DM, DM);

    // 6) LN2
    ln_kernel<<<Mrows, 256, 0, stream>>>(x2, ln2_g, ln2_b, nullptr, h2, 0);

    // 7) MLP up + exact GELU: [16384 x 1024] x [1024 x 4096]
    gemm_bf16_wmma<1><<<dim3((4 * DM) / 128, Mrows / 256), 256, 0, stream>>>(
        h2, w1T, b1, nullptr, nullptr, mid, Mrows, 4 * DM, DM);

    // 8) MLP down + residual -> d_out (fp32): [16384 x 4096] x [4096 x 1024]
    gemm_bf16_wmma<3><<<dim3(DM / 128, Mrows / 256), 256, 0, stream>>>(
        mid, w2T, b2, x2, nullptr, d_out, Mrows, DM, 4 * DM);
}
